// Trans_token_SSG_33655363732085
// MI455X (gfx1250) — matrix operand order repaired and checked by reading, not compiled, not run on hardware
//
#include <hip/hip_runtime.h>
#include <hip/hip_bf16.h>

typedef __attribute__((ext_vector_type(16))) _Float16 v16h;
typedef __attribute__((ext_vector_type(8)))  float    v8f;

#define BQ 8
#define NN 2048
#define DM 128
#define HEADS 4
#define HD 32
#define DFF 512
#define DOUT 256
#define SPTS 512
#define EPSV 1e-5f

// ---------------- WMMA fragment helpers (wave32, contiguous k = (L>>4)*16 + i) ----------

// A 16x32 f16 fragment. src points at tile origin, row-major, leading dim ld.
// lane L: row = L&15 ; halves 0..15 -> K = (L>>4)*16 + i.
__device__ inline v16h load_a_frag(const _Float16* __restrict__ src, int ld) {
  int lane = threadIdx.x & 31;
  int r = lane & 15;
  int kb = (lane >> 4) << 4;
  const _Float16* p = src + r * ld + kb;
  v16h a;
#pragma unroll
  for (int i = 0; i < 8; ++i) a[i] = p[i];
#pragma unroll
  for (int i = 0; i < 8; ++i) a[8 + i] = p[8 + i];
  return a;
}

// B 32x16 f16 fragment where B[k][n] = w[n*ld + k] (source holds B^T row-major,
// e.g. a weight matrix W[Nd][K] or a K-tile [keys][hd]). w points at (col0,k0).
__device__ inline v16h load_b_fragT(const _Float16* __restrict__ w, int ld) {
  int lane = threadIdx.x & 31;
  int n = lane & 15;
  int kb = (lane >> 4) << 4;
  const _Float16* p = w + n * ld + kb;
  v16h b;
#pragma unroll
  for (int i = 0; i < 16; ++i) b[i] = p[i];
  return b;
}

// B 32x16 f16 fragment where B[k][n] = src[k*ld + n] (row-major [K][N], e.g. a
// V tile: rows = keys, cols = head dims). src points at (k0, col0).
__device__ inline v16h load_b_frag(const _Float16* __restrict__ src, int ld) {
  int lane = threadIdx.x & 31;
  int n = lane & 15;
  int kb = (lane >> 4) << 4;
  const _Float16* p = src + kb * ld + n;
  v16h b;
#pragma unroll
  for (int i = 0; i < 16; ++i) b[i] = p[i * ld];
  return b;
}

__device__ inline v8f wmma_f16(v16h a, v16h b, v8f c) {
  return __builtin_amdgcn_wmma_f32_16x16x32_f16(false, a, false, b,
                                                (short)0, c, false, false);
}

// gfx1250 async global->LDS copy, 16B per lane, tracked by ASYNCcnt.
// GV addressing form: vdst = per-lane LDS byte offset, vaddr = 64-bit address.
__device__ inline void async_cp16(void* lds, const void* gptr) {
  unsigned int loff = (unsigned int)(unsigned long long)lds; // low 32 = LDS offset
  asm volatile("global_load_async_to_lds_b128 %0, %1, off"
               :: "v"(loff), "v"((unsigned long long)gptr) : "memory");
}
__device__ inline void wait_async0() {
  asm volatile("s_wait_asynccnt 0" ::: "memory");
}

// ---------------- small elementwise kernels ----------------------------------

__global__ void k_f32_to_f16(const float* __restrict__ s, _Float16* __restrict__ d, int n) {
  int i = blockIdx.x * blockDim.x + threadIdx.x;
  if (i < n) d[i] = (_Float16)s[i];
}

// points [B,Cin,N] -> ptsT f16 [B,N,Cin]
__global__ void k_transpose_pts(const float* __restrict__ pts, _Float16* __restrict__ d,
                                int B, int C, int N) {
  int i = blockIdx.x * blockDim.x + threadIdx.x;
  if (i >= B * N * C) return;
  int c = i % C; int t = i / C; int n = t % N; int b = t / N;
  d[i] = (_Float16)pts[((size_t)b * C + c) * N + n];
}

// x = BN(feat) + pos
__global__ void k_make_x(const float* __restrict__ feat, const float* __restrict__ xyz,
                         const float* __restrict__ Wpos, const float* __restrict__ bpos,
                         const float* __restrict__ g, const float* __restrict__ bt,
                         const float* __restrict__ mu, const float* __restrict__ vr,
                         float* __restrict__ xf, _Float16* __restrict__ xh, int B, int N) {
  int i = blockIdx.x * blockDim.x + threadIdx.x;
  if (i >= B * N * DM) return;
  int d = i & (DM - 1); int row = i >> 7; int n = row % N; int b = row / N;
  float f = feat[i];
  f = (f - mu[d]) * rsqrtf(vr[d] + EPSV) * g[d] + bt[d];
  const float* xz = xyz + (size_t)b * 3 * N + n;
  float pos = bpos[d] + Wpos[d * 3 + 0] * xz[0] + Wpos[d * 3 + 1] * xz[N]
            + Wpos[d * 3 + 2] * xz[2 * N];
  float x = f + pos;
  xf[i] = x;
  xh[i] = (_Float16)x;
}

// y = LN(res + delta); one wave per 128-wide row
__global__ void k_ln_res(const float* __restrict__ res, const float* __restrict__ delta,
                         const float* __restrict__ g, const float* __restrict__ bt,
                         float* __restrict__ yf, _Float16* __restrict__ yh, int rows) {
  int row = blockIdx.x * (blockDim.x >> 5) + (threadIdx.x >> 5);
  if (row >= rows) return;
  int lane = threadIdx.x & 31;
  float v[4], s = 0.f, sq = 0.f;
#pragma unroll
  for (int j = 0; j < 4; ++j) {
    int d = lane + j * 32;
    float t = res[row * DM + d] + delta[row * DM + d];
    v[j] = t; s += t; sq += t * t;
  }
#pragma unroll
  for (int mk = 1; mk < 32; mk <<= 1) {
    s += __shfl_xor(s, mk, 32);
    sq += __shfl_xor(sq, mk, 32);
  }
  float mean = s * (1.f / DM);
  float varr = sq * (1.f / DM) - mean * mean;
  float inv = rsqrtf(varr + EPSV);
#pragma unroll
  for (int j = 0; j < 4; ++j) {
    int d = lane + j * 32;
    float y = (v[j] - mean) * inv * g[d] + bt[d];
    if (yf) yf[row * DM + d] = y;
    if (yh) yh[row * DM + d] = (_Float16)y;
  }
}

// -------- WMMA GEMM: C[M,Nd] = A[M,K]*W[Nd,K]^T, 1x4 tile register blocking ---

__global__ void k_gemm_nt(const _Float16* __restrict__ A, const _Float16* __restrict__ W,
                          const float* __restrict__ bias, float* __restrict__ Cf,
                          _Float16* __restrict__ Ch, int M, int Nd, int K, int relu) {
  int wave = blockIdx.x * (blockDim.x >> 5) + (threadIdx.x >> 5);
  int grp = Nd >> 6;                          // 4 n-tiles per wave (Nd % 64 == 0)
  int total = (M >> 4) * grp;
  if (wave >= total) return;
  int tm = wave / grp, tg = wave % grp;
  const _Float16* abase = A + (size_t)(tm * 16) * K;
  v8f acc[4] = {{}, {}, {}, {}};
  for (int k0 = 0; k0 < K; k0 += 32) {
    v16h a = load_a_frag(abase + k0, K);      // A frag reused for 4 WMMAs
#pragma unroll
    for (int j = 0; j < 4; ++j) {
      v16h bf = load_b_fragT(W + (size_t)(tg * 64 + j * 16) * K + k0, K);
      acc[j] = wmma_f16(a, bf, acc[j]);
    }
  }
  int lane = threadIdx.x & 31;
  int n = lane & 15, rbse = (lane >> 4) << 3;
#pragma unroll
  for (int j = 0; j < 4; ++j) {
    int col = tg * 64 + j * 16 + n;
    float bb = bias ? bias[col] : 0.f;
#pragma unroll
    for (int r = 0; r < 8; ++r) {
      int row = tm * 16 + rbse + r;
      float v = acc[j][r] + bb;
      if (relu) v = fmaxf(v, 0.f);
      if (Cf) Cf[(size_t)row * Nd + col] = v;
      if (Ch) Ch[(size_t)row * Nd + col] = (_Float16)v;
    }
  }
}

// -------- flash attention: block = 4 waves sharing (b,h); async-LDS K/V -------
// Each block handles 4 consecutive 16-query tiles. 32x32 K and V tiles are
// DMA'd into LDS (double-buffered, ASYNCcnt) and shared by all 4 waves.

__global__ void k_flash_attn(const _Float16* __restrict__ qh, const _Float16* __restrict__ kh,
                             const _Float16* __restrict__ vh, _Float16* __restrict__ oh,
                             int B, int N) {
  __shared__ _Float16 Kt[2][32][32];
  __shared__ _Float16 Vt[2][32][32];
  __shared__ _Float16 Pl[4][16][32];
  int wv = threadIdx.x >> 5;
  int lane = threadIdx.x & 31;
  int qgrps = N >> 6;                        // query-tile groups of 4
  int g = blockIdx.x;
  int qg = g % qgrps; int bh = g / qgrps;
  int h = bh % HEADS; int b = bh / HEADS;
  int qt = (qg << 2) + wv;
  int row0 = b * N + qt * 16;
  int n = lane & 15, rb = (lane >> 4) << 3;

  v16h aq = load_a_frag(qh + (size_t)row0 * DM + h * HD, DM);   // Q tile 16x32
  v8f o0 = {}, o1 = {};
  float m[8], l[8];
#pragma unroll
  for (int r = 0; r < 8; ++r) { m[r] = -1e30f; l[r] = 0.f; }
  const float scale = 0.1767766952966369f;   // 1/sqrt(32)

  int t = threadIdx.x;
  int srow = t >> 2, scol = (t & 3) << 3;    // 128 lanes x 16B cover a 32x32 tile
  int nkt = N >> 5;
  {                                          // preload tile 0
    int kb = b * N;
    async_cp16(&Kt[0][srow][scol], kh + (size_t)(kb + srow) * DM + h * HD + scol);
    async_cp16(&Vt[0][srow][scol], vh + (size_t)(kb + srow) * DM + h * HD + scol);
  }
  int buf = 0;
  for (int kt = 0; kt < nkt; ++kt) {
    wait_async0();
    __syncthreads();                         // tiles[buf] visible to all waves
    if (kt + 1 < nkt) {                      // overlap next DMA with compute
      int kb2 = b * N + (kt + 1) * 32;
      async_cp16(&Kt[buf ^ 1][srow][scol], kh + (size_t)(kb2 + srow) * DM + h * HD + scol);
      async_cp16(&Vt[buf ^ 1][srow][scol], vh + (size_t)(kb2 + srow) * DM + h * HD + scol);
    }
    const _Float16* kbase = &Kt[buf][0][0];
    const _Float16* vbase = &Vt[buf][0][0];
    // S = Q*K^T : B[k(hd)][key] = Kt[key][hd]
    v16h b0 = load_b_fragT(kbase, 32);
    v16h b1 = load_b_fragT(kbase + 16 * 32, 32);
    v8f z = {};
    v8f s0 = wmma_f16(aq, b0, z);
    v8f s1 = wmma_f16(aq, b1, z);
#pragma unroll
    for (int r = 0; r < 8; ++r) {
      float a0 = s0[r] * scale, a1 = s1[r] * scale;
      float mx = fmaxf(a0, a1);
      mx = fmaxf(mx, __shfl_xor(mx, 1, 32));
      mx = fmaxf(mx, __shfl_xor(mx, 2, 32));
      mx = fmaxf(mx, __shfl_xor(mx, 4, 32));
      mx = fmaxf(mx, __shfl_xor(mx, 8, 32));
      float mn = fmaxf(m[r], mx);
      float corr = __expf(m[r] - mn);
      float p0 = __expf(a0 - mn), p1 = __expf(a1 - mn);
      float rs = p0 + p1;
      rs += __shfl_xor(rs, 1, 32);
      rs += __shfl_xor(rs, 2, 32);
      rs += __shfl_xor(rs, 4, 32);
      rs += __shfl_xor(rs, 8, 32);
      l[r] = l[r] * corr + rs;
      o0[r] *= corr; o1[r] *= corr;
      m[r] = mn;
      Pl[wv][rb + r][n]      = (_Float16)p0;
      Pl[wv][rb + r][16 + n] = (_Float16)p1;
    }
    __builtin_amdgcn_wave_barrier();
    v16h ap  = load_a_frag(&Pl[wv][0][0], 32);   // P 16x32 (C->A layout relay)
    v16h bv0 = load_b_frag(vbase, 32);           // V keys x hd0..15
    v16h bv1 = load_b_frag(vbase + 16, 32);      // hd16..31
    o0 = wmma_f16(ap, bv0, o0);
    o1 = wmma_f16(ap, bv1, o1);
    __syncthreads();                         // all reads of buf done before reuse
    buf ^= 1;
  }
#pragma unroll
  for (int r = 0; r < 8; ++r) {
    float inv = 1.f / l[r];
    int row = row0 + rb + r;
    oh[(size_t)row * DM + h * HD + n]      = (_Float16)(o0[r] * inv);
    oh[(size_t)row * DM + h * HD + 16 + n] = (_Float16)(o1[r] * inv);
  }
}

// ---------------- FPS (serial tail): 1 block per batch -----------------------

__global__ void k_fps(const float* __restrict__ xyz, int* __restrict__ idx,
                      int B, int N, int S) {
  __shared__ float dist[NN];
  __shared__ float sval[256];
  __shared__ int   sidx[256];
  __shared__ float cent[3];
  int b = blockIdx.x, tid = threadIdx.x;
  const float* xb = xyz + (size_t)b * 3 * N;
  for (int n = tid; n < N; n += 256) dist[n] = 1e10f;
  int far = 0;
  for (int j = 0; j < S; ++j) {
    if (tid == 0) idx[b * S + j] = far;
    if (tid < 3) cent[tid] = xb[tid * N + far];
    __syncthreads();
    float cx = cent[0], cy = cent[1], cz = cent[2];
    float best = -1.f; int bi = 0;
    for (int n = tid; n < N; n += 256) {
      float dx = xb[n] - cx, dy = xb[N + n] - cy, dz = xb[2 * N + n] - cz;
      float d = dx * dx + dy * dy + dz * dz;
      float nd = fminf(dist[n], d);
      dist[n] = nd;
      if (nd > best) { best = nd; bi = n; }
    }
    sval[tid] = best; sidx[tid] = bi;
    __syncthreads();
    for (int st = 128; st > 0; st >>= 1) {
      if (tid < st) {
        float ov = sval[tid + st]; int oi = sidx[tid + st];
        if (ov > sval[tid] || (ov == sval[tid] && oi < sidx[tid])) {
          sval[tid] = ov; sidx[tid] = oi;   // first-max tie-break (jnp.argmax)
        }
      }
      __syncthreads();
    }
    far = sidx[0];
    __syncthreads();
  }
}

// gather: write new_xyz output + build cat rows [features | xyz] (stride 132)
__global__ void k_gather(const float* __restrict__ xyz, const float* __restrict__ feat,
                         const int* __restrict__ idx, float* __restrict__ outxyz,
                         float* __restrict__ cat, int B, int N, int S) {
  int bs = blockIdx.x; int s = bs % S; int b = bs / S;
  int id = idx[b * S + s];
  int tid = threadIdx.x;   // 128
  float* crow = cat + (size_t)bs * 132;
  crow[tid] = feat[((size_t)b * N + id) * DM + tid];
  if (tid < 3) {
    float c = xyz[((size_t)b * 3 + tid) * N + id];
    crow[DM + tid] = c;
    outxyz[((size_t)b * 3 + tid) * S + s] = c;
  }
  if (tid == 127) crow[131] = 0.f;
}

// final FC + BN + relu : out[b, o, s]
__global__ void k_fc(const float* __restrict__ cat, const float* __restrict__ Wfc,
                     const float* __restrict__ bfc, const float* __restrict__ g,
                     const float* __restrict__ bt, const float* __restrict__ mu,
                     const float* __restrict__ vr, float* __restrict__ out, int B, int S) {
  __shared__ float row[132];
  int bs = blockIdx.x; int s = bs % S; int b = bs / S;
  int o = threadIdx.x;   // 256
  if (o < 132) row[o] = cat[(size_t)bs * 132 + o];
  __syncthreads();
  float acc = bfc[o];
  const float* w = Wfc + o * 131;
  for (int d = 0; d < 131; ++d) acc += w[d] * row[d];
  acc = (acc - mu[o]) * rsqrtf(vr[o] + EPSV) * g[o] + bt[o];
  acc = fmaxf(acc, 0.f);
  out[((size_t)b * DOUT + o) * S + s] = acc;
}

// ---------------- host orchestration -----------------------------------------

extern "C" void kernel_launch(void* const* d_in, const int* in_sizes, int n_in,
                              void* d_out, int out_size, void* d_ws, size_t ws_size,
                              hipStream_t stream) {
  const float* xyz    = (const float*)d_in[0];
  const float* points = (const float*)d_in[1];
  const float* W_proj = (const float*)d_in[2];
  const float* b_proj = (const float*)d_in[3];
  const float* bn1_g  = (const float*)d_in[4];
  const float* bn1_b  = (const float*)d_in[5];
  const float* bn1_m  = (const float*)d_in[6];
  const float* bn1_v  = (const float*)d_in[7];
  const float* W_pos  = (const float*)d_in[8];
  const float* b_pos  = (const float*)d_in[9];
  const float* Wq = (const float*)d_in[10]; const float* bq = (const float*)d_in[11];
  const float* Wk = (const float*)d_in[12]; const float* bk = (const float*)d_in[13];
  const float* Wv = (const float*)d_in[14]; const float* bv = (const float*)d_in[15];
  const float* Wo = (const float*)d_in[16]; const float* bo = (const float*)d_in[17];
  const float* ln1_g = (const float*)d_in[18]; const float* ln1_b = (const float*)d_in[19];
  const float* W1 = (const float*)d_in[20]; const float* b1 = (const float*)d_in[21];
  const float* W2 = (const float*)d_in[22]; const float* b2 = (const float*)d_in[23];
  const float* ln2_g = (const float*)d_in[24]; const float* ln2_b = (const float*)d_in[25];
  const float* W_fc = (const float*)d_in[26]; const float* b_fc = (const float*)d_in[27];
  const float* bn2_g = (const float*)d_in[28]; const float* bn2_b = (const float*)d_in[29];
  const float* bn2_m = (const float*)d_in[30]; const float* bn2_v = (const float*)d_in[31];

  const int B = BQ, N = NN, Cin = 64, D = DM, Dff = DFF, S = SPTS;
  const int BN = B * N;   // 16384

  char* ws = (char*)d_ws;
  size_t off = 0;
  auto alloc = [&](size_t bytes) -> void* {
    void* p = ws + off;
    off = (off + bytes + 255) & ~(size_t)255;
    return p;
  };

  _Float16* wproj_h = (_Float16*)alloc((size_t)D * Cin * 2);
  _Float16* wq_h = (_Float16*)alloc((size_t)D * D * 2);
  _Float16* wk_h = (_Float16*)alloc((size_t)D * D * 2);
  _Float16* wv_h = (_Float16*)alloc((size_t)D * D * 2);
  _Float16* wo_h = (_Float16*)alloc((size_t)D * D * 2);
  _Float16* w1_h = (_Float16*)alloc((size_t)Dff * D * 2);
  _Float16* w2_h = (_Float16*)alloc((size_t)D * Dff * 2);
  _Float16* ptsT_h = (_Float16*)alloc((size_t)BN * Cin * 2);
  float*    tmpf = (float*)alloc((size_t)BN * D * 4);      // proj / o-proj / ffn2 out
  float*    xf   = (float*)alloc((size_t)BN * D * 4);      // x, later features
  _Float16* xh   = (_Float16*)alloc((size_t)BN * D * 2);   // x f16, later x1 f16
  _Float16* qh   = (_Float16*)alloc((size_t)BN * D * 2);
  _Float16* kh   = (_Float16*)alloc((size_t)BN * D * 2);
  _Float16* vh   = (_Float16*)alloc((size_t)BN * D * 2);
  _Float16* ah   = (_Float16*)alloc((size_t)BN * D * 2);   // attention out
  float*    x1f  = (float*)alloc((size_t)BN * D * 4);
  _Float16* h1h  = (_Float16*)alloc((size_t)BN * Dff * 2); // ffn hidden
  int*      fidx = (int*)alloc((size_t)B * S * 4);
  float*    catf = (float*)alloc((size_t)B * S * 132 * 4);

  auto cvt = [&](const float* s, _Float16* d, int n) {
    k_f32_to_f16<<<(n + 255) / 256, 256, 0, stream>>>(s, d, n);
  };
  cvt(W_proj, wproj_h, D * Cin);
  cvt(Wq, wq_h, D * D); cvt(Wk, wk_h, D * D);
  cvt(Wv, wv_h, D * D); cvt(Wo, wo_h, D * D);
  cvt(W1, w1_h, Dff * D); cvt(W2, w2_h, D * Dff);

  k_transpose_pts<<<(BN * Cin + 255) / 256, 256, 0, stream>>>(points, ptsT_h, B, Cin, N);

  auto gemm = [&](const _Float16* A, const _Float16* W, const float* bias,
                  float* Cf, _Float16* Ch, int M, int Nd, int K, int relu) {
    int grps = (M >> 4) * (Nd >> 6);
    k_gemm_nt<<<(grps + 3) / 4, 128, 0, stream>>>(A, W, bias, Cf, Ch, M, Nd, K, relu);
  };

  // projection: feat = ptsT * W_proj^T + b_proj   (BN x 128, K=64)
  gemm(ptsT_h, wproj_h, b_proj, tmpf, nullptr, BN, D, Cin, 0);
  // x = BN(feat) + pos
  k_make_x<<<(BN * D + 255) / 256, 256, 0, stream>>>(tmpf, xyz, W_pos, b_pos,
      bn1_g, bn1_b, bn1_m, bn1_v, xf, xh, B, N);
  // QKV
  gemm(xh, wq_h, bq, nullptr, qh, BN, D, D, 0);
  gemm(xh, wk_h, bk, nullptr, kh, BN, D, D, 0);
  gemm(xh, wv_h, bv, nullptr, vh, BN, D, D, 0);
  // attention: block = 4 waves share (b,h), 4 q-tiles; 1024 blocks
  {
    int blocks = B * HEADS * (N >> 6);
    k_flash_attn<<<blocks, 128, 0, stream>>>(qh, kh, vh, ah, B, N);
  }
  // o-proj + residual LN1  -> x1 (f32) and x1 f16 (reuse xh)
  gemm(ah, wo_h, bo, tmpf, nullptr, BN, D, D, 0);
  k_ln_res<<<(BN + 3) / 4, 128, 0, stream>>>(xf, tmpf, ln1_g, ln1_b, x1f, xh, BN);
  // FFN
  gemm(xh, w1_h, b1, nullptr, h1h, BN, Dff, D, 1);     // relu
  gemm(h1h, w2_h, b2, tmpf, nullptr, BN, D, Dff, 0);
  // LN2 -> features (reuse xf)
  k_ln_res<<<(BN + 3) / 4, 128, 0, stream>>>(x1f, tmpf, ln2_g, ln2_b, xf, nullptr, BN);
  // FPS
  k_fps<<<B, 256, 0, stream>>>(xyz, fidx, B, N, S);
  // gather: writes new_xyz part of d_out and cat rows
  float* out_xyz = (float*)d_out;                 // [B,3,S]
  float* out_fc  = out_xyz + (size_t)B * 3 * S;   // [B,256,S]
  k_gather<<<B * S, 128, 0, stream>>>(xyz, xf, fidx, out_xyz, catf, B, N, S);
  // final FC + BN + relu
  k_fc<<<B * S, 256, 0, stream>>>(catf, W_fc, b_fc, bn2_g, bn2_b, bn2_m, bn2_v,
                                  out_fc, B, S);
}
